// SBSignalBranch_28587302323140
// MI455X (gfx1250) — matrix operand order repaired and checked by reading, not compiled, not run on hardware
//
#include <hip/hip_runtime.h>

#define BD 128
#define LAYERS 8

typedef __attribute__((ext_vector_type(16))) __bf16          v16bf;
typedef __attribute__((ext_vector_type(16))) unsigned short  v16us;
typedef __attribute__((ext_vector_type(8)))  float           v8f;

static __device__ __forceinline__ unsigned short f2bf(float f) {
  unsigned u = __builtin_bit_cast(unsigned, f);
  u += 0x7FFFu + ((u >> 16) & 1u);          // round-to-nearest-even
  return (unsigned short)(u >> 16);
}

// Raw hardware TRANS ops (no IEEE refinement sequences).
static __device__ __forceinline__ float fexp(float x) {      // e^x
  return __builtin_amdgcn_exp2f(x * 1.4426950408889634f);
}
static __device__ __forceinline__ float frcp(float x) {
  return __builtin_amdgcn_rcpf(x);
}
static __device__ __forceinline__ float fsigmoid(float x) {  // 1/(1+e^-x)
  return frcp(1.f + fexp(-x));
}
static __device__ __forceinline__ float ftanh(float x) {     // 1 - 2/(e^2x+1)
  return 1.f - 2.f * frcp(fexp(2.f * x) + 1.f);
}

// ---------------------------------------------------------------------------
// Pre-pass: convert Wg/Wp (f32, [L,256,128]) into bf16 B-matrix fragments laid
// out exactly as v_wmma_f32_16x16x32_bf16 wants them:
//   fragment (l,k,n): lane ld, slot e ->  K = k*32 + (ld>=16?8:0) + (e<8?e:e+8),
//                                         N = n*16 + (ld&15)
// Each fragment = 512 bf16 = 1 KB; a lane's 16 halves are 32 contiguous bytes.
// ---------------------------------------------------------------------------
__global__ void convert_weights(const float* __restrict__ Wg,
                                const float* __restrict__ Wp,
                                unsigned short* __restrict__ wsG,
                                unsigned short* __restrict__ wsP) {
  int idx = blockIdx.x * blockDim.x + threadIdx.x;   // 0 .. L*8*8*512-1
  int e    = idx & 15;
  int ldn  = (idx >> 4) & 31;
  int frag = idx >> 9;
  int n = frag & 7;
  int k = (frag >> 3) & 7;
  int l = frag >> 6;
  int K = k * 32 + ((ldn >= 16) ? 8 : 0) + ((e < 8) ? e : e + 8);
  int N = n * 16 + (ldn & 15);
  int widx = (l * 256 + K) * 128 + N;
  wsG[idx] = f2bf(Wg[widx]);
  wsP[idx] = f2bf(Wp[widx]);
}

// ---------------------------------------------------------------------------
// Main kernel: one wave32 per 16-row tile, one wave per block.
// LDS (32 KB): cur f32 | prev f32 | acc f32 | cur A-frags bf16 | prev A-frags bf16
// A-fragment staging layout: afrag[k][lane][slot] (k = 0..3), so the K-loop
// reads a lane's v16bf with two ds_load_b128 and does zero VALU work.
// ---------------------------------------------------------------------------
__global__ __launch_bounds__(32) void adaptive_step_kernel(
    const float* __restrict__ signal, const float* __restrict__ previous,
    const float* __restrict__ bp, const float* __restrict__ bg,
    const float* __restrict__ Ws, const float* __restrict__ bs,
    const float* __restrict__ gamma, const float* __restrict__ beta,
    const unsigned short* __restrict__ wsG,
    const unsigned short* __restrict__ wsP,
    float* __restrict__ out, int B) {
  __shared__ __attribute__((aligned(32))) float          ldsF[3 * 2048];
  __shared__ __attribute__((aligned(32))) unsigned short ldsA[2 * 2048];

  const int lane   = threadIdx.x & 31;
  const int lane15 = lane & 15;
  const int hi     = lane >> 4;           // which 16-lane half
  const int tile   = blockIdx.x;

  float* cur  = &ldsF[0];
  float* prv  = cur + 2048;
  float* accm = prv + 2048;
  unsigned short* curA = &ldsA[0];        // 4 ksteps x 512 bf16
  unsigned short* prvA = curA + 2048;

  const float* gsig = signal   + (size_t)tile * 16 * BD;
  const float* gprv = previous + (size_t)tile * 16 * BD;

  // ---- load tile state into wave-private LDS (f32 copies) ----
#pragma unroll
  for (int i = 0; i < 16; ++i) {
    int idx = lane + i * 32;
    ((float4*)cur)[idx] = ((const float4*)gsig)[idx];
    ((float4*)prv)[idx] = ((const float4*)gprv)[idx];
    float4 z; z.x = z.y = z.z = z.w = 0.f;
    ((float4*)accm)[idx] = z;
  }

  // ---- build bf16 A-fragment staging for signal (cur) and previous ----
  // A-layout: lane ld, slot e -> (M = ld&15, K = k*32 + hi*8 + (e<8 ? e : e+8))
#pragma unroll
  for (int k = 0; k < 4; ++k) {
    const float* rpc = gsig + lane15 * BD + k * 32 + hi * 8;
    const float* rpp = gprv + lane15 * BD + k * 32 + hi * 8;
#pragma unroll
    for (int half = 0; half < 2; ++half) {   // half 0: cur, 1: prev
      const float* rp = half ? rpp : rpc;
      float4 a0 = *(const float4*)(rp);
      float4 a1 = *(const float4*)(rp + 4);
      float4 a2 = *(const float4*)(rp + 16);
      float4 a3 = *(const float4*)(rp + 20);
      v16us au;
      au[0]  = f2bf(a0.x); au[1]  = f2bf(a0.y); au[2]  = f2bf(a0.z); au[3]  = f2bf(a0.w);
      au[4]  = f2bf(a1.x); au[5]  = f2bf(a1.y); au[6]  = f2bf(a1.z); au[7]  = f2bf(a1.w);
      au[8]  = f2bf(a2.x); au[9]  = f2bf(a2.y); au[10] = f2bf(a2.z); au[11] = f2bf(a2.w);
      au[12] = f2bf(a3.x); au[13] = f2bf(a3.y); au[14] = f2bf(a3.z); au[15] = f2bf(a3.w);
      unsigned short* dst = (half ? prvA : curA) + k * 512 + lane * 16;
      *(v16us*)dst = au;
    }
  }

  // per-row state, rows M = j + hi*8, replicated across the 16 lanes of a half
  float active[8], depth[8], smass[8], gmacc[8], dmacc[8], aqacc[8];
#pragma unroll
  for (int j = 0; j < 8; ++j) {
    active[j] = 1.f; depth[j] = 0.f; smass[j] = 0.f;
    gmacc[j] = 0.f; dmacc[j] = 0.f; aqacc[j] = 0.f;
  }

  for (int l = 0; l < LAYERS; ++l) {
    float s_part[8], q_part[8], gm_part[8];
#pragma unroll
    for (int j = 0; j < 8; ++j) { s_part[j] = 0.f; q_part[j] = 0.f; gm_part[j] = 0.f; }
    v8f pre[8];

    // ---- two fused GEMMs: [16,256] x [256,128] for gate and candidate ----
#pragma unroll
    for (int n = 0; n < 8; ++n) {
      v8f accG, accP;
#pragma unroll
      for (int j = 0; j < 8; ++j) { accG[j] = 0.f; accP[j] = 0.f; }
#pragma unroll
      for (int k = 0; k < 8; ++k) {
        const unsigned short* ap =
            ((k < 4) ? curA : prvA) + (k & 3) * 512 + lane * 16;
        v16bf A = __builtin_bit_cast(v16bf, *(const v16us*)ap);

        const int frag = (((l * 8 + k) * 8 + n) << 9) + lane * 16;
        v16bf Bg = __builtin_bit_cast(v16bf, *(const v16us*)(wsG + frag));
        v16bf Bp = __builtin_bit_cast(v16bf, *(const v16us*)(wsP + frag));

        accG = __builtin_amdgcn_wmma_f32_16x16x32_bf16(false, A, false, Bg,
                                                       (short)0, accG, false, false);
        accP = __builtin_amdgcn_wmma_f32_16x16x32_bf16(false, A, false, Bp,
                                                       (short)0, accP, false, false);
      }
      // elementwise in C/D layout: element (M = j + hi*8, N = col)
      const int col  = n * 16 + lane15;
      const float bgv = bg[l * BD + col];
      const float bpv = bp[l * BD + col];
#pragma unroll
      for (int j = 0; j < 8; ++j) {
        float g  = fsigmoid(accG[j] + bgv);
        float c  = ftanh(accP[j] + bpv);
        float cv = cur[(j + hi * 8) * BD + col];
        float x  = cv + g * (c - cv);
        pre[n][j] = x;
        s_part[j]  += x;
        q_part[j]  += x * x;
        gm_part[j] += g;
      }
    }

    // ---- row reductions (stay inside each 16-lane half: masks 1,2,4,8) ----
#pragma unroll
    for (int m = 1; m < 16; m <<= 1) {
#pragma unroll
      for (int j = 0; j < 8; ++j) {
        s_part[j]  += __shfl_xor(s_part[j],  m, 32);
        q_part[j]  += __shfl_xor(q_part[j],  m, 32);
        gm_part[j] += __shfl_xor(gm_part[j], m, 32);
      }
    }
    float mu[8], inv[8];
#pragma unroll
    for (int j = 0; j < 8; ++j) {
      mu[j] = s_part[j] * (1.f / 128.f);
      float var = q_part[j] * (1.f / 128.f) - mu[j] * mu[j];
      inv[j] = __builtin_amdgcn_rsqf(var + 1e-5f);
    }

    // ---- layernorm + delta/adequacy partials; write cur (f32 + bf16 frags) ----
    float d_part[8], a_part[8];
#pragma unroll
    for (int j = 0; j < 8; ++j) { d_part[j] = 0.f; a_part[j] = 0.f; }
#pragma unroll
    for (int n = 0; n < 8; ++n) {
      const int col  = n * 16 + lane15;
      const float gv   = gamma[l * BD + col];
      const float bv   = beta[l * BD + col];
      const float wsu  = Ws[l * 256 + col];
      const float wspv = Ws[l * 256 + 128 + col];
      // A-fragment scatter coordinates for column `col` (K index of joined):
      //   kstep = n>>1 ; hi' = lane15>>3 ; slot e = (lane15&7) + (n&1)*8
      const int abase = (n >> 1) * 512 + (lane15 & 7) + (n & 1) * 8 +
                        ((lane15 >> 3) * 16 + hi * 8) * 16;
#pragma unroll
      for (int j = 0; j < 8; ++j) {
        float u  = (pre[n][j] - mu[j]) * inv[j] * gv + bv;
        const int a = (j + hi * 8) * BD + col;
        float cv = cur[a];
        float dd = u - cv;
        d_part[j] += dd * dd;
        float pv = prv[a];
        a_part[j] += u * wsu + pv * wspv;
        pre[n][j] = u;                    // now holds `updated`
        cur[a] = u;                       // current <- updated (f32)
        curA[abase + j * 16] = f2bf(u);   // current <- updated (bf16 A-frag)
      }
    }
#pragma unroll
    for (int m = 1; m < 16; m <<= 1) {
#pragma unroll
      for (int j = 0; j < 8; ++j) {
        d_part[j] += __shfl_xor(d_part[j], m, 32);
        a_part[j] += __shfl_xor(a_part[j], m, 32);
      }
    }

    // ---- halting bookkeeping (per row) ----
    const float bsl = bs[l];
    float sm[8];
#pragma unroll
    for (int j = 0; j < 8; ++j) {
      float adequacy = fsigmoid(a_part[j] + bsl);
      float sp       = fsigmoid((adequacy - 0.6f) * 10.f);
      sm[j]     = active[j] * sp;
      depth[j] += sm[j] * (float)(l + 1);
      smass[j] += sm[j];
      active[j] *= (1.f - sp);
      aqacc[j] += adequacy;
      dmacc[j] += __builtin_amdgcn_sqrtf(d_part[j]);
      gmacc[j] += gm_part[j] * (1.f / 128.f);
    }

    // ---- acc += stop_mass * updated ----
#pragma unroll
    for (int n = 0; n < 8; ++n) {
      const int col = n * 16 + lane15;
#pragma unroll
      for (int j = 0; j < 8; ++j) {
        const int a = (j + hi * 8) * BD + col;
        accm[a] += sm[j] * pre[n][j];
      }
    }
  }

  // ---- epilogue ----
  const size_t SOFF = (size_t)B * BD;
#pragma unroll
  for (int n = 0; n < 8; ++n) {
    const int col = n * 16 + lane15;
#pragma unroll
    for (int j = 0; j < 8; ++j) {
      const int a   = (j + hi * 8) * BD + col;
      const int row = tile * 16 + j + hi * 8;
      out[(size_t)row * BD + col] = accm[a] + active[j] * cur[a];
    }
  }
  if (lane15 == 0) {
#pragma unroll
    for (int j = 0; j < 8; ++j) {
      const size_t row = (size_t)tile * 16 + j + hi * 8;
      out[SOFF + row]                 = gmacc[j] * (1.f / (float)LAYERS);
      out[SOFF + (size_t)B + row]     = dmacc[j] * (1.f / (float)LAYERS);
      out[SOFF + 2 * (size_t)B + row] = depth[j] + active[j] * (float)LAYERS;
      out[SOFF + 3 * (size_t)B + row] = smass[j];
      out[SOFF + 4 * (size_t)B + row] = aqacc[j] * (1.f / (float)LAYERS);
    }
  }
}

// ---------------------------------------------------------------------------
extern "C" void kernel_launch(void* const* d_in, const int* in_sizes, int n_in,
                              void* d_out, int out_size, void* d_ws, size_t ws_size,
                              hipStream_t stream) {
  const float* signal   = (const float*)d_in[0];
  const float* previous = (const float*)d_in[1];
  const float* Wp       = (const float*)d_in[2];
  const float* bp       = (const float*)d_in[3];
  const float* Wg       = (const float*)d_in[4];
  const float* bg       = (const float*)d_in[5];
  const float* Ws       = (const float*)d_in[6];
  const float* bs       = (const float*)d_in[7];
  const float* gamma    = (const float*)d_in[8];
  const float* beta     = (const float*)d_in[9];
  float* out = (float*)d_out;

  const int B = in_sizes[0] / BD;

  const int frag_elems = LAYERS * 8 * 8 * 512;            // 262144 bf16 per matmul
  unsigned short* wsG = (unsigned short*)d_ws;
  unsigned short* wsP = wsG + frag_elems;

  convert_weights<<<frag_elems / 256, 256, 0, stream>>>(Wg, Wp, wsG, wsP);

  const int tiles = B / 16;                                // 16 rows per wave
  adaptive_step_kernel<<<tiles, 32, 0, stream>>>(
      signal, previous, bp, bg, Ws, bs, gamma, beta, wsG, wsP, out, B);
}
